// sLSTMblock_42399917146725
// MI455X (gfx1250) — compile-verified, matrix operand order checked by reading, not run on hardware
//
#include <hip/hip_runtime.h>
#include <hip/hip_bf16.h>
#include <math.h>

// ---------------------------------------------------------------------------
// sLSTM block for MI455X (gfx1250): WMMA f16 GEMMs for conv + gates,
// LDS-resident input window for the causal conv, row-LN reduction kernels.
// ---------------------------------------------------------------------------

typedef __attribute__((ext_vector_type(16))) _Float16 v16h;
typedef __attribute__((ext_vector_type(8)))  _Float16 v8h;
typedef __attribute__((ext_vector_type(8)))  float    v8f;

#define Dm    512
#define KC    64
#define MTOT  16384      // B*S = 8*2048
#define SEQ   2048
#define DFFv  682
#define EPSv  1e-5f
#define LSTR  520        // padded LDS row stride in halves (1040B, 16B aligned)

// ---------------- block reduction (works for 256 or 1024 threads) ----------
__device__ __forceinline__ float block_sum(float v, float* sbuf) {
#pragma unroll
  for (int off = 16; off > 0; off >>= 1) v += __shfl_down(v, off, 32);
  int lane = threadIdx.x & 31;
  int wid  = threadIdx.x >> 5;
  int nw   = blockDim.x >> 5;
  if (lane == 0) sbuf[wid] = v;
  __syncthreads();
  float t = (threadIdx.x < (unsigned)nw) ? sbuf[threadIdx.x] : 0.0f;
  if (wid == 0) {
#pragma unroll
    for (int off = 16; off > 0; off >>= 1) t += __shfl_down(t, off, 32);
    if (lane == 0) sbuf[0] = t;
  }
  __syncthreads();
  float r = sbuf[0];
  __syncthreads();
  return r;
}

// LN over a 512-wide row, 256 threads x 2 elements each
__device__ __forceinline__ void ln2(float x0, float x1,
                                    const float* g, const float* b,
                                    int d0, int d1, float* red,
                                    float& y0, float& y1) {
  float mu  = block_sum(x0 + x1, red) * (1.0f / 512.0f);
  float var = block_sum(x0 * x0 + x1 * x1, red) * (1.0f / 512.0f) - mu * mu;
  float inv = rsqrtf(var + EPSv);
  y0 = (x0 - mu) * inv * g[d0] + b[d0];
  y1 = (x1 - mu) * inv * g[d1] + b[d1];
}

// ---------------- WMMA fragment load from LDS (16-bit A/B layout) ----------
// lanes 0-15: idx = lane,     K halves {kb..kb+7, 16+kb..16+kb+7}, kb=0
// lanes16-31: idx = lane-16,  same with kb=8
__device__ __forceinline__ v16h load_frag(const _Float16* rowbase, int kb) {
  v8h lo = *(const v8h*)(rowbase + kb);
  v8h hi = *(const v8h*)(rowbase + 16 + kb);
  return __builtin_shufflevector(lo, hi, 0, 1, 2, 3, 4, 5, 6, 7,
                                 8, 9, 10, 11, 12, 13, 14, 15);
}

// ---------------- prep kernels --------------------------------------------
__global__ void k_zero(float* Csum, float* Nsum, float* Hsum) {
  int t = threadIdx.x;
  Csum[t] = 0.f; Nsum[t] = 0.f; Hsum[t] = 0.f;
}

// conv_w [n][c][k] f32  ->  Wt [k][n][c] f16
__global__ void k_prep_convw(const float* __restrict__ Cw, _Float16* __restrict__ Wt) {
  int idx = blockIdx.x * 256 + threadIdx.x;          // 512*512*64 total
  int c = idx & 511, n = (idx >> 9) & 511, k = idx >> 18;
  Wt[idx] = (_Float16)Cw[((size_t)n * 512 + c) * 64 + k];
}

// W [in c][out n] f32 -> Wt [n][c] f16
__global__ void k_prep_w(const float* __restrict__ W, _Float16* __restrict__ Wt) {
  int idx = blockIdx.x * 256 + threadIdx.x;          // 512*512 total
  int c = idx & 511, n = idx >> 9;
  Wt[(size_t)n * 512 + c] = (_Float16)W[(size_t)c * 512 + n];
}

// LN(x) -> xn (f16); one block per row, 256 thr x 2 elems
__global__ void k_ln_rows(const float* __restrict__ X,
                          const float* __restrict__ g, const float* __restrict__ b,
                          _Float16* __restrict__ Out) {
  __shared__ float red[32];
  int m = blockIdx.x, d0 = threadIdx.x, d1 = threadIdx.x + 256;
  const float* row = X + (size_t)m * Dm;
  float x0 = row[d0], x1 = row[d1], y0, y1;
  ln2(x0, x1, g, b, d0, d1, red, y0, y1);
  Out[(size_t)m * Dm + d0] = (_Float16)y0;
  Out[(size_t)m * Dm + d1] = (_Float16)y1;
}

// ---------------- causal conv: LDS-resident window, 64 taps of WMMA --------
// out[m,n] = silu( sum_k sum_c xn[m+k-63, c] * Wt[k][n][c] + cb[n] )
// Dynamic LDS: xwin[95][LSTR] (window rows t0-63..t0+31) + bk[64][LSTR].
__global__ void k_conv_wmma(const _Float16* __restrict__ Xn,
                            const _Float16* __restrict__ Wt,
                            const float* __restrict__ cb,
                            _Float16* __restrict__ Xc) {
  extern __shared__ __align__(16) char smem_raw[];
  _Float16* xwin = (_Float16*)smem_raw;              // 95 * LSTR halves
  _Float16* bk   = xwin + 95 * LSTR;                 // 64 * LSTR halves
  int m0 = blockIdx.x * 32, n0 = blockIdx.y * 64;
  int bb = m0 >> 11, t0 = m0 & 2047;
  int tid = threadIdx.x, lane = tid & 31, wid = tid >> 5;
  int wrow = wid >> 2, wcol = wid & 3;
  int fm = lane & 15, kb = (lane < 16) ? 0 : 8;

  // stage the whole shifted input window once (reused by all 64 taps)
  for (int i = tid; i < 95 * 64; i += 256) {
    int w = i >> 6, seg = i & 63;
    int t = t0 + w - (KC - 1);
    v8h val;
    if (t >= 0)
      val = *(const v8h*)(Xn + ((size_t)(bb * SEQ + t) * Dm + seg * 8));
    else {
#pragma unroll
      for (int j = 0; j < 8; ++j) val[j] = (_Float16)0.0f;
    }
    *(v8h*)(xwin + w * LSTR + seg * 8) = val;
  }

  v8f acc = {};
  for (int k = 0; k < KC; ++k) {
    __syncthreads();                                  // bk reuse fence
    // prefetch next tap's weight tile into L2 while we work on this one
    if (k + 1 < KC)
      __builtin_prefetch(Wt + ((size_t)(k + 1) * Dm + n0 + (tid & 63)) * Dm, 0, 1);
    // stage B tile for tap k: 64 n-rows x 512 c
    for (int i = tid; i < 64 * 64; i += 256) {
      int n = i >> 6, seg = i & 63;
      *(v8h*)(bk + n * LSTR + seg * 8) =
          *(const v8h*)(Wt + ((size_t)k * Dm + n0 + n) * Dm + seg * 8);
    }
    __syncthreads();
    const _Float16* arow = xwin + (wrow * 16 + fm + k) * LSTR;  // shifted row
    const _Float16* brow = bk + (wcol * 16 + fm) * LSTR;
#pragma unroll
    for (int cc = 0; cc < 16; ++cc) {
      v16h a = load_frag(arow + cc * 32, kb);
      v16h b = load_frag(brow + cc * 32, kb);
      acc = __builtin_amdgcn_wmma_f32_16x16x32_f16(false, a, false, b,
                                                   (short)0, acc, false, false);
    }
  }
  int mB = m0 + wrow * 16 + ((lane >> 4) << 3);
  int nI = n0 + wcol * 16 + (lane & 15);
  float bn = cb[nI];
#pragma unroll
  for (int r = 0; r < 8; ++r) {
    float v = acc[r] + bn;
    float s = v / (1.0f + expf(-v));                 // SiLU
    Xc[(size_t)(mB + r) * Dm + nI] = (_Float16)s;
  }
}

// ---------------- dense WMMA GEMM: out = A[MT,512] * Bt[n][k]^T + bias -----
// Dynamic LDS: at[32][LSTR] + bt[64][LSTR]; single stage, 16 unrolled WMMAs.
__global__ void k_gemm_wmma(const _Float16* __restrict__ A,
                            const _Float16* __restrict__ Bt,
                            const float* __restrict__ bias,
                            float* __restrict__ Cout) {
  extern __shared__ __align__(16) char smem_raw[];
  _Float16* at = (_Float16*)smem_raw;                // 32 * LSTR halves
  _Float16* bt = at + 32 * LSTR;                     // 64 * LSTR halves
  int m0 = blockIdx.x * 32, n0 = blockIdx.y * 64;
  int tid = threadIdx.x, lane = tid & 31, wid = tid >> 5;
  int wrow = wid >> 2, wcol = wid & 3;
  int fm = lane & 15, kb = (lane < 16) ? 0 : 8;

  for (int i = tid; i < 32 * 64; i += 256) {         // stage full-K A tile
    int row = i >> 6, seg = i & 63;
    *(v8h*)(at + row * LSTR + seg * 8) =
        *(const v8h*)(A + (size_t)(m0 + row) * Dm + seg * 8);
  }
  for (int i = tid; i < 64 * 64; i += 256) {         // stage full-K B tile
    int n = i >> 6, seg = i & 63;
    *(v8h*)(bt + n * LSTR + seg * 8) =
        *(const v8h*)(Bt + (size_t)(n0 + n) * Dm + seg * 8);
  }
  __syncthreads();

  v8f acc = {};
  const _Float16* arow = at + (wrow * 16 + fm) * LSTR;
  const _Float16* brow = bt + (wcol * 16 + fm) * LSTR;
#pragma unroll
  for (int cc = 0; cc < 16; ++cc) {
    v16h a = load_frag(arow + cc * 32, kb);
    v16h b = load_frag(brow + cc * 32, kb);
    acc = __builtin_amdgcn_wmma_f32_16x16x32_f16(false, a, false, b,
                                                 (short)0, acc, false, false);
  }
  int mB = m0 + wrow * 16 + ((lane >> 4) << 3);
  int nI = n0 + wcol * 16 + (lane & 15);
  float bn = bias[nI];
#pragma unroll
  for (int r = 0; r < 8; ++r)
    Cout[(size_t)(mB + r) * Dm + nI] = acc[r] + bn;
}

// ---------------- gate nonlinearity + per-row LN + mean accumulation -------
__global__ void k_gates(const float* __restrict__ AI, const float* __restrict__ AF,
                        const float* __restrict__ AO, const float* __restrict__ AZ,
                        const float* gi, const float* bi, const float* gf, const float* bf,
                        const float* go, const float* bo, const float* gz, const float* bz,
                        const float* gc, const float* bc, const float* gn, const float* bn,
                        float* __restrict__ Csum, float* __restrict__ Nsum,
                        float* __restrict__ OB) {
  __shared__ float red[32];
  int m = blockIdx.x, d0 = threadIdx.x, d1 = threadIdx.x + 256;
  size_t off = (size_t)m * Dm;
  float li0, li1, lf0, lf1, lo0, lo1, lz0, lz1;
  ln2(AI[off + d0], AI[off + d1], gi, bi, d0, d1, red, li0, li1);
  ln2(AF[off + d0], AF[off + d1], gf, bf, d0, d1, red, lf0, lf1);
  ln2(AO[off + d0], AO[off + d1], go, bo, d0, d1, red, lo0, lo1);
  ln2(AZ[off + d0], AZ[off + d1], gz, bz, d0, d1, red, lz0, lz1);
  // max-stabilized exp gating (mt_1 = 0)
  float i0 = expf(li0 - fmaxf(lf0, li0));
  float i1 = expf(li1 - fmaxf(lf1, li1));
  float o0 = 1.0f / (1.0f + expf(-lo0));
  float o1 = 1.0f / (1.0f + expf(-lo1));
  float z0 = tanhf(lz0), z1 = tanhf(lz1);
  float ct0 = i0 * z0, ct1 = i1 * z1;                // ct_1 = 0
  float nt0 = i0, nt1 = i1;                          // nt_1 = 0
  float c0, c1, n0v, n1v;
  ln2(ct0, ct1, gc, bc, d0, d1, red, c0, c1);
  ln2(nt0, nt1, gn, bn, d0, d1, red, n0v, n1v);
  atomicAdd(&Csum[d0], c0); atomicAdd(&Csum[d1], c1);
  atomicAdd(&Nsum[d0], n0v); atomicAdd(&Nsum[d1], n1v);
  OB[off + d0] = o0; OB[off + d1] = o1;
}

__global__ void k_ratio(const float* Csum, const float* Nsum, float* ratio) {
  int t = threadIdx.x;
  ratio[t] = Csum[t] / Nsum[t];                      // means: 1/M cancels
}

// ht = o * (C/N); per-row LN(ln_h); accumulate mean
__global__ void k_ht(const float* __restrict__ OB, const float* __restrict__ ratio,
                     const float* gh, const float* bh, float* __restrict__ Hsum) {
  __shared__ float red[32];
  int m = blockIdx.x, d0 = threadIdx.x, d1 = threadIdx.x + 256;
  size_t off = (size_t)m * Dm;
  float h0 = OB[off + d0] * ratio[d0];
  float h1 = OB[off + d1] * ratio[d1];
  float y0, y1;
  ln2(h0, h1, gh, bh, d0, d1, red, y0, y1);
  atomicAdd(&Hsum[d0], y0); atomicAdd(&Hsum[d1], y1);
}

// ---------------- head: gn-LN, Wl/Wr + exact GELU, ln_out-LN, Wp -----------
__global__ void k_head(const float* __restrict__ Hsum,
                       const float* gng, const float* gnb,
                       const float* __restrict__ Wl, const float* __restrict__ bl,
                       const float* __restrict__ Wr, const float* __restrict__ br,
                       const float* log, const float* lob,
                       const float* __restrict__ Wp, const float* __restrict__ bp,
                       float* __restrict__ out) {
  __shared__ float red[32];
  __shared__ float sv[Dm];
  __shared__ float pv[DFFv];
  int t = threadIdx.x;
  float hv = (t < Dm) ? Hsum[t] * (1.0f / (float)MTOT) : 0.0f;
  float mu  = block_sum(hv, red) * (1.0f / 512.0f);
  float var = block_sum(hv * hv, red) * (1.0f / 512.0f) - mu * mu;
  if (t < Dm) sv[t] = (hv - mu) * rsqrtf(var + EPSv) * gng[t] + gnb[t];
  __syncthreads();
  float pval = 0.0f;
  if (t < DFFv) {
    float l = bl[t], r = br[t];
    for (int d = 0; d < Dm; ++d) {
      float sd = sv[d];
      l += sd * Wl[(size_t)d * DFFv + t];
      r += sd * Wr[(size_t)d * DFFv + t];
    }
    float gr = 0.5f * r * (1.0f + erff(r * 0.70710678118f));  // exact GELU
    pval = l * gr;
  }
  float mu2  = block_sum(pval, red) * (1.0f / (float)DFFv);
  float var2 = block_sum(pval * pval, red) * (1.0f / (float)DFFv) - mu2 * mu2;
  if (t < DFFv) pv[t] = (pval - mu2) * rsqrtf(var2 + EPSv) * log[t] + lob[t];
  __syncthreads();
  if (t < Dm) {
    float acc = bp[t];
    for (int j = 0; j < DFFv; ++j) acc += pv[j] * Wp[(size_t)j * Dm + t];
    out[t] = acc;
  }
}

// ---------------------------------------------------------------------------
extern "C" void kernel_launch(void* const* d_in, const int* in_sizes, int n_in,
                              void* d_out, int out_size, void* d_ws, size_t ws_size,
                              hipStream_t stream) {
  (void)in_sizes; (void)n_in; (void)out_size; (void)ws_size;
  const float* x      = (const float*)d_in[0];
  const float* conv_w = (const float*)d_in[1];
  const float* conv_b = (const float*)d_in[2];
  const float* Wi = (const float*)d_in[3],  *bi = (const float*)d_in[4];
  const float* Wf = (const float*)d_in[6],  *bf = (const float*)d_in[7];
  const float* Wo = (const float*)d_in[9],  *bo = (const float*)d_in[10];
  const float* Wz = (const float*)d_in[12], *bz = (const float*)d_in[13];
  const float* ln_g = (const float*)d_in[15],  *ln_b = (const float*)d_in[16];
  const float* li_g = (const float*)d_in[17],  *li_b = (const float*)d_in[18];
  const float* lf_g = (const float*)d_in[19],  *lf_b = (const float*)d_in[20];
  const float* lo_g = (const float*)d_in[21],  *lo_b = (const float*)d_in[22];
  const float* lz_g = (const float*)d_in[23],  *lz_b = (const float*)d_in[24];
  const float* gn_g = (const float*)d_in[25],  *gn_b = (const float*)d_in[26];
  const float* lc_g = (const float*)d_in[27],  *lc_b = (const float*)d_in[28];
  const float* lnn_g = (const float*)d_in[29], *lnn_b = (const float*)d_in[30];
  const float* lh_g = (const float*)d_in[31],  *lh_b = (const float*)d_in[32];
  const float* lout_g = (const float*)d_in[33], *lout_b = (const float*)d_in[34];
  const float* Wl = (const float*)d_in[35], *bl = (const float*)d_in[36];
  const float* Wr = (const float*)d_in[37], *br = (const float*)d_in[38];
  const float* Wp = (const float*)d_in[39], *bp = (const float*)d_in[40];
  float* out = (float*)d_out;

  // workspace layout
  char* ws = (char*)d_ws;
  _Float16* xn16 = (_Float16*)(ws);                       // 16 MB
  _Float16* xc16 = (_Float16*)(ws + (size_t)16777216);    // 16 MB
  _Float16* wtc  = (_Float16*)(ws + (size_t)33554432);    // 32 MB  [k][n][c]
  _Float16* wtg  = (_Float16*)(ws + (size_t)67108864);    // 4 x 0.5 MB [n][c]
  float* ai   = (float*)(ws + (size_t)69206016);          // 32 MB each
  float* af   = (float*)(ws + (size_t)69206016 + 33554432ull);
  float* ao   = (float*)(ws + (size_t)69206016 + 2ull * 33554432ull);
  float* az   = (float*)(ws + (size_t)69206016 + 3ull * 33554432ull);
  float* obuf = (float*)(ws + (size_t)203423744);         // 32 MB
  float* Csum = (float*)(ws + (size_t)236978176);
  float* Nsum = Csum + 512;
  float* Hsum = Csum + 1024;
  float* ratio = Csum + 1536;

  k_zero<<<1, 512, 0, stream>>>(Csum, Nsum, Hsum);
  k_ln_rows<<<MTOT, 256, 0, stream>>>(x, ln_g, ln_b, xn16);
  k_prep_convw<<<(512 * 512 * 64) / 256, 256, 0, stream>>>(conv_w, wtc);
  k_prep_w<<<1024, 256, 0, stream>>>(Wi, wtg + 0ull * 262144);
  k_prep_w<<<1024, 256, 0, stream>>>(Wf, wtg + 1ull * 262144);
  k_prep_w<<<1024, 256, 0, stream>>>(Wo, wtg + 2ull * 262144);
  k_prep_w<<<1024, 256, 0, stream>>>(Wz, wtg + 3ull * 262144);

  dim3 gg(MTOT / 32, Dm / 64, 1);
  size_t shConv = (size_t)(95 * LSTR + 64 * LSTR) * sizeof(_Float16);  // 165,360 B
  size_t shGemm = (size_t)(32 * LSTR + 64 * LSTR) * sizeof(_Float16);  //  99,840 B
  k_conv_wmma<<<gg, 256, shConv, stream>>>(xn16, wtc, conv_b, xc16);
  k_gemm_wmma<<<gg, 256, shGemm, stream>>>(xc16, wtg + 0ull * 262144, bi, ai);
  k_gemm_wmma<<<gg, 256, shGemm, stream>>>(xc16, wtg + 1ull * 262144, bf, af);
  k_gemm_wmma<<<gg, 256, shGemm, stream>>>(xn16, wtg + 2ull * 262144, bo, ao);
  k_gemm_wmma<<<gg, 256, shGemm, stream>>>(xn16, wtg + 3ull * 262144, bz, az);

  k_gates<<<MTOT, 256, 0, stream>>>(ai, af, ao, az,
                                    li_g, li_b, lf_g, lf_b, lo_g, lo_b, lz_g, lz_b,
                                    lc_g, lc_b, lnn_g, lnn_b, Csum, Nsum, obuf);
  k_ratio<<<1, 512, 0, stream>>>(Csum, Nsum, ratio);
  k_ht<<<MTOT, 256, 0, stream>>>(obuf, ratio, lh_g, lh_b, Hsum);
  k_head<<<1, 1024, 0, stream>>>(Hsum, gn_g, gn_b, Wl, bl, Wr, br,
                                 lout_g, lout_b, Wp, bp, out);
}
